// MambaEncoder_8684423873293
// MI455X (gfx1250) — matrix operand order repaired and checked by reading, not compiled, not run on hardware
//
#include <hip/hip_runtime.h>
#include <math.h>
#include <cstdint>

// ---------------------------------------------------------------------------
// Problem constants (from the reference)
// ---------------------------------------------------------------------------
#define BB   4
#define TT   1024
#define INF_ 256
#define HH   512
#define LL   4
#define DI   1024          // 2*H
#define SS   16
#define RR   32            // H/16
#define KK   4
#define OUTF 256
#define NTOK (BB*TT)       // 4096 rows
#define RMS_EPS 1.1920929e-07f
#define LN_EPS  1e-05f

// ---------------------------------------------------------------------------
// WMMA types (CDNA5 / gfx1250, wave32)
// ---------------------------------------------------------------------------
typedef __attribute__((ext_vector_type(16))) __bf16 v16bf;
typedef __attribute__((ext_vector_type(8)))  float  v8f;
typedef unsigned short bf16_t;   // bf16 bit pattern

static __device__ __forceinline__ bf16_t f32_to_bf16_rne(float f) {
  unsigned x = __float_as_uint(f);
  if ((x & 0x7FFFFFFFu) > 0x7F800000u) return (bf16_t)0x7FC0;   // NaN
  unsigned r = (x + 0x7FFFu + ((x >> 16) & 1u)) >> 16;          // RNE
  return (bf16_t)r;
}

static __device__ __forceinline__ float sigmoidf_(float x) { return 1.0f / (1.0f + __expf(-x)); }
static __device__ __forceinline__ float siluf_(float x)    { return x * sigmoidf_(x); }
static __device__ __forceinline__ float softplusf_(float x){ return x > 20.0f ? x : log1pf(__expf(x)); }

// ---------------------------------------------------------------------------
// bf16-input / f32-accumulate WMMA GEMM with async (ASYNCcnt) LDS staging:
//   C[M,N] = epilogue( A[M,K] * W[N,K]^T ),  A,W already bf16 in global memory
// Block tile 64x64, 256 threads (8 waves), each wave -> 16(M) x 32(N).
// Double-buffered tiles moved with GLOBAL_LOAD_ASYNC_TO_LDS_B128 (one b128 per
// thread per tile), completion via s_wait_asynccnt (in-order) + barrier.
// LDS rows padded to 80B: 16B-aligned for b128, and 20-uint stride gives
// conflict-free ds reads for the 16-row fragment pattern (gcd(20,64)=4,
// 20*r mod 64 distinct for r=0..15).
// EPI: 0 plain, 1 +bias, 2 (+bias)*rowscale[m], 3 softplus(+bias), 4 +C (resid)
// DUALOUT: additionally write a bf16 mirror of C into Cb.
// ---------------------------------------------------------------------------
#define LDSW 20                      // uints per 64-byte row (+16B pad)
#define TILEB (64 * LDSW * 4)        // 5120 bytes per tile buffer

template <int EPI, bool DUALOUT>
__global__ __launch_bounds__(256)
void gemm_bf16_wmma(const bf16_t* __restrict__ A, int lda,
                    const bf16_t* __restrict__ W, int ldw,
                    float* __restrict__ C, int ldc,
                    bf16_t* __restrict__ Cb,
                    int K,
                    const float* __restrict__ bias,
                    const float* __restrict__ rowscale) {
  __shared__ __align__(16) unsigned As[2][64 * LDSW];
  __shared__ __align__(16) unsigned Ws[2][64 * LDSW];

  const int tid   = threadIdx.x;
  const int lane  = tid & 31;
  const int wave  = tid >> 5;
  const int waveM = wave >> 1;       // 0..3
  const int waveN = wave & 1;        // 0..1
  const int m0    = blockIdx.x * 64;
  const int n0    = blockIdx.y * 64;

  v8f acc0 = {};
  v8f acc1 = {};

  // tile-copy mapping: thread -> one 16B chunk (8 bf16) of each 64x32 tile
  const int rowLd = tid >> 2;          // 0..63
  const int kLd8  = (tid & 3) * 8;     // element offset within row: 0,8,16,24

  const unsigned ldsA = (unsigned)(unsigned long long)(uintptr_t)&As[0][0];
  const unsigned ldsW = (unsigned)(unsigned long long)(uintptr_t)&Ws[0][0];
  const unsigned chunk = (unsigned)(rowLd * (LDSW * 4) + (tid & 3) * 16);

  const int half = lane >> 4;          // 0 or 1
  const int l16  = lane & 15;

  auto issue_tiles = [&](int kt, int buf) {
    const unsigned dA = ldsA + (unsigned)buf * TILEB + chunk;
    const unsigned dW = ldsW + (unsigned)buf * TILEB + chunk;
    const unsigned long long gA =
        (unsigned long long)(uintptr_t)(A + (size_t)(m0 + rowLd) * lda + kt + kLd8);
    const unsigned long long gW =
        (unsigned long long)(uintptr_t)(W + (size_t)(n0 + rowLd) * ldw + kt + kLd8);
    asm volatile("global_load_async_to_lds_b128 %0, %1, off"
                 :: "v"(dA), "v"(gA) : "memory");
    asm volatile("global_load_async_to_lds_b128 %0, %1, off"
                 :: "v"(dW), "v"(gW) : "memory");
  };

  const int nk = K >> 5;
  issue_tiles(0, 0);

  for (int i = 0; i < nk; ++i) {
    const int cur = i & 1;
    if (i + 1 < nk) {
      issue_tiles((i + 1) << 5, cur ^ 1);
      asm volatile("s_wait_asynccnt 0x2" ::: "memory");   // older pair retired
    } else {
      asm volatile("s_wait_asynccnt 0x0" ::: "memory");
    }
    __syncthreads();

    // A fragment: 16-bit A 16x32 layout (ISA 7.12.2):
    // lane m=l16, half h: VGPR j<4 -> K pair (8h+2j); j>=4 -> K pair (16+8h+2(j-4))
    union { v16bf v; unsigned u[8]; } fa, fb0, fb1;
    const int rowA = waveM * 16 + l16;
#pragma unroll
    for (int j = 0; j < 4; ++j) {
      fa.u[j]     = As[cur][rowA * LDSW + 4 * half + j];
      fa.u[j + 4] = As[cur][rowA * LDSW + 8 + 4 * half + j];
    }
    // B fragment (32x16): same K order as A: half h, VGPR j<4 -> K pair (8h+2j);
    // j>=4 -> K pair (16+8h+2(j-4)).
    const int rowB0 = waveN * 32 + l16;
    const int rowB1 = rowB0 + 16;
#pragma unroll
    for (int j = 0; j < 8; ++j) {
      fb0.u[j] = Ws[cur][rowB0 * LDSW + 4 * half + j + 4 * (j >> 2)];
      fb1.u[j] = Ws[cur][rowB1 * LDSW + 4 * half + j + 4 * (j >> 2)];
    }

    acc0 = __builtin_amdgcn_wmma_f32_16x16x32_bf16(false, fa.v, false, fb0.v,
                                                   (short)0, acc0, false, false);
    acc1 = __builtin_amdgcn_wmma_f32_16x16x32_bf16(false, fa.v, false, fb1.v,
                                                   (short)0, acc1, false, false);
    __syncthreads();
  }

  // Epilogue. C/D f32 layout: lane l16 = N, VGPR v -> M = v + 8*half.
  const int nBase0 = n0 + waveN * 32 + l16;
  const int nBase1 = nBase0 + 16;
#pragma unroll
  for (int v = 0; v < 8; ++v) {
    const int m = m0 + waveM * 16 + v + 8 * half;
    float r0 = acc0[v];
    float r1 = acc1[v];
    if (EPI == 1 || EPI == 2 || EPI == 3) { r0 += bias[nBase0]; r1 += bias[nBase1]; }
    if (EPI == 2) { const float s = rowscale[m]; r0 *= s; r1 *= s; }
    if (EPI == 3) { r0 = softplusf_(r0); r1 = softplusf_(r1); }
    if (EPI == 4) {
      r0 += C[(size_t)m * ldc + nBase0];
      r1 += C[(size_t)m * ldc + nBase1];
    }
    C[(size_t)m * ldc + nBase0] = r0;
    C[(size_t)m * ldc + nBase1] = r1;
    if (DUALOUT) {
      Cb[(size_t)m * ldc + nBase0] = f32_to_bf16_rne(r0);
      Cb[(size_t)m * ldc + nBase1] = f32_to_bf16_rne(r1);
    }
  }
}

// ---------------------------------------------------------------------------
// f32 -> bf16 bulk weight conversion (once per launch)
// ---------------------------------------------------------------------------
__global__ __launch_bounds__(256)
void convert_bf16_kernel(const float* __restrict__ in, bf16_t* __restrict__ out, int n) {
  const int i = blockIdx.x * 256 + threadIdx.x;
  if (i < n) out[i] = f32_to_bf16_rne(in[i]);
}

// ---------------------------------------------------------------------------
// Row NaN mask + nan_to_num + bf16 conversion of x. One wave per row (IN=256).
// ---------------------------------------------------------------------------
__global__ __launch_bounds__(256)
void rowmask_conv_kernel(const float* __restrict__ x, float* __restrict__ mask,
                         bf16_t* __restrict__ xb) {
  const int lane = threadIdx.x & 31;
  const int row  = blockIdx.x * 8 + (threadIdx.x >> 5);
  unsigned any = 0;
  const float* xr = x + (size_t)row * INF_;
  bf16_t* xo = xb + (size_t)row * INF_;
#pragma unroll
  for (int j = lane; j < INF_; j += 32) {
    float v = xr[j];
    const unsigned isn = (v != v) ? 1u : 0u;
    any |= isn;
    if (isn) v = 0.0f;
    xo[j] = f32_to_bf16_rne(v);
  }
#pragma unroll
  for (int off = 16; off >= 1; off >>= 1) any |= __shfl_xor(any, off, 32);
  if (lane == 0) mask[row] = any ? 0.0f : 1.0f;
}

// ---------------------------------------------------------------------------
// RMSNorm over H=512, one wave per row -> bf16 output (feeds in_proj GEMM)
// ---------------------------------------------------------------------------
__global__ __launch_bounds__(256)
void rmsnorm_kernel(const float* __restrict__ h, const float* __restrict__ w,
                    bf16_t* __restrict__ xnb) {
  const int lane = threadIdx.x & 31;
  const int row  = blockIdx.x * 8 + (threadIdx.x >> 5);
  const float* hr = h + (size_t)row * HH;
  float s = 0.0f;
#pragma unroll
  for (int j = lane; j < HH; j += 32) { const float v = hr[j]; s += v * v; }
#pragma unroll
  for (int off = 16; off >= 1; off >>= 1) s += __shfl_xor(s, off, 32);
  const float scale = rsqrtf(s * (1.0f / HH) + RMS_EPS);
  bf16_t* xo = xnb + (size_t)row * HH;
#pragma unroll
  for (int j = lane; j < HH; j += 32) xo[j] = f32_to_bf16_rne(hr[j] * scale * w[j]);
}

// ---------------------------------------------------------------------------
// LayerNorm over H=512, one wave per row -> bf16 output (feeds final GEMM)
// ---------------------------------------------------------------------------
__global__ __launch_bounds__(256)
void layernorm_kernel(const float* __restrict__ h, const float* __restrict__ g,
                      const float* __restrict__ b, bf16_t* __restrict__ hnb) {
  const int lane = threadIdx.x & 31;
  const int row  = blockIdx.x * 8 + (threadIdx.x >> 5);
  const float* hr = h + (size_t)row * HH;
  float s1 = 0.0f, s2 = 0.0f;
#pragma unroll
  for (int j = lane; j < HH; j += 32) { const float v = hr[j]; s1 += v; s2 += v * v; }
#pragma unroll
  for (int off = 16; off >= 1; off >>= 1) {
    s1 += __shfl_xor(s1, off, 32);
    s2 += __shfl_xor(s2, off, 32);
  }
  const float mu  = s1 * (1.0f / HH);
  const float var = s2 * (1.0f / HH) - mu * mu;
  const float inv = rsqrtf(var + LN_EPS);
  bf16_t* ho = hnb + (size_t)row * HH;
#pragma unroll
  for (int j = lane; j < HH; j += 32)
    ho[j] = f32_to_bf16_rne((hr[j] - mu) * inv * g[j] + b[j]);
}

// ---------------------------------------------------------------------------
// Depthwise causal conv (K=4) over time + bias + SiLU:
//   u[b,t,c] = silu( sum_k cw[c,k]*xx[b,t-3+k,c] + cb[c] ),  xx = xz[...,:Di]
// writes u (f32, for the scan) and ub (bf16, for the x_proj GEMM)
// ---------------------------------------------------------------------------
__global__ __launch_bounds__(256)
void conv_silu_kernel(const float* __restrict__ xz, const float* __restrict__ cw,
                      const float* __restrict__ cb, float* __restrict__ u,
                      bf16_t* __restrict__ ub) {
  const int bt = blockIdx.x;             // 0..4095
  const int t  = bt & (TT - 1);
  const int c  = blockIdx.y * 256 + threadIdx.x;
  float acc = cb[c];
#pragma unroll
  for (int k = 0; k < KK; ++k) {
    const int ttime = t - (KK - 1) + k;
    if (ttime >= 0) acc += cw[c * KK + k] * xz[(size_t)(bt - (KK - 1) + k) * (2 * DI) + c];
  }
  const float v = siluf_(acc);
  u[(size_t)bt * DI + c]  = v;
  ub[(size_t)bt * DI + c] = f32_to_bf16_rne(v);
}

// ---------------------------------------------------------------------------
// Selective scan. One lane per (b, channel d); 16 states in registers.
//   st[s] = st[s]*exp(delta*A[d,s]) + (delta*u)*Bm[t,s];  y = sum st[s]*Cm[t,s]
//   out   = (y + u*Dp[d]) * silu(z)   -> bf16 (feeds out-proj GEMM)
// grid = B * (Di/256) = 16 blocks of 256 threads
// ---------------------------------------------------------------------------
__global__ __launch_bounds__(256)
void scan_kernel(const float* __restrict__ delta, const float* __restrict__ u,
                 const float* __restrict__ xd, const float* __restrict__ xz,
                 const float* __restrict__ A_log, const float* __restrict__ Dp,
                 bf16_t* __restrict__ ygb) {
  __shared__ float sBC[32];              // B[0..15], C[16..31] for current t
  const int bidx = blockIdx.x >> 2;      // batch
  const int d    = (blockIdx.x & 3) * 256 + threadIdx.x;

  float a[SS];
#pragma unroll
  for (int s = 0; s < SS; ++s) a[s] = -__expf(A_log[(size_t)d * SS + s]);
  const float dpd = Dp[d];

  float st[SS];
#pragma unroll
  for (int s = 0; s < SS; ++s) st[s] = 0.0f;

  for (int t = 0; t < TT; ++t) {
    const size_t row = (size_t)bidx * TT + t;
    if (threadIdx.x < 32) sBC[threadIdx.x] = xd[row * 64 + RR + threadIdx.x];
    __syncthreads();

    const float dl = delta[row * DI + d];
    const float ut = u[row * DI + d];
    const float du = dl * ut;
    float y = 0.0f;
#pragma unroll
    for (int s = 0; s < SS; ++s) {
      st[s] = st[s] * __expf(dl * a[s]) + du * sBC[s];
      y += st[s] * sBC[16 + s];
    }
    const float z = xz[row * (2 * DI) + DI + d];
    ygb[row * DI + d] = f32_to_bf16_rne((y + ut * dpd) * siluf_(z));
    __syncthreads();
  }
}

// ---------------------------------------------------------------------------
// Host-side orchestration
// ---------------------------------------------------------------------------
extern "C" void kernel_launch(void* const* d_in, const int* in_sizes, int n_in,
                              void* d_out, int out_size, void* d_ws, size_t ws_size,
                              hipStream_t stream) {
  (void)in_sizes; (void)n_in; (void)out_size; (void)ws_size;

  const float* x        = (const float*)d_in[0];   // (B,T,IN)
  const float* in_w     = (const float*)d_in[1];   // (H,IN)
  const float* in_b     = (const float*)d_in[2];   // (H)
  const float* rms_w    = (const float*)d_in[3];   // (L,H)
  const float* in_proj  = (const float*)d_in[4];   // (L,2Di,H)
  const float* conv_w   = (const float*)d_in[5];   // (L,Di,K)
  const float* conv_b   = (const float*)d_in[6];   // (L,Di)
  const float* x_proj   = (const float*)d_in[7];   // (L,64,Di)
  const float* dt_w     = (const float*)d_in[8];   // (L,Di,R)
  const float* dt_b     = (const float*)d_in[9];   // (L,Di)
  const float* A_log    = (const float*)d_in[10];  // (L,Di,S)
  const float* D_ssm    = (const float*)d_in[11];  // (L,Di)
  const float* mb_out   = (const float*)d_in[12];  // (L,H,Di)
  const float* ln_g     = (const float*)d_in[13];  // (H)
  const float* ln_b     = (const float*)d_in[14];  // (H)
  const float* out_w    = (const float*)d_in[15];  // (OUT,H)
  const float* out_b    = (const float*)d_in[16];  // (OUT)
  float*       outp     = (float*)d_out;           // (B,T,OUT)

  // workspace layout
  char*  ws  = (char*)d_ws;
  size_t off = 0;
  auto allocb = [&](size_t bytes) -> void* {
    void* p = (void*)(ws + off);
    off += (bytes + 255) & ~(size_t)255;
    return p;
  };
  float*  mask  = (float*) allocb((size_t)NTOK * 4);
  bf16_t* xb    = (bf16_t*)allocb((size_t)NTOK * INF_ * 2);
  float*  h     = (float*) allocb((size_t)NTOK * HH * 4);
  bf16_t* xnb   = (bf16_t*)allocb((size_t)NTOK * HH * 2);
  float*  xz    = (float*) allocb((size_t)NTOK * 2 * DI * 4);
  float*  u     = (float*) allocb((size_t)NTOK * DI * 4);
  bf16_t* ub    = (bf16_t*)allocb((size_t)NTOK * DI * 2);
  float*  xd    = (float*) allocb((size_t)NTOK * 64 * 4);
  bf16_t* xdb   = (bf16_t*)allocb((size_t)NTOK * 64 * 2);
  float*  delta = (float*) allocb((size_t)NTOK * DI * 4);
  bf16_t* ygb   = (bf16_t*)allocb((size_t)NTOK * DI * 2);
  bf16_t* hnb   = (bf16_t*)allocb((size_t)NTOK * HH * 2);
  // bf16 weights
  bf16_t* in_wb  = (bf16_t*)allocb((size_t)HH * INF_ * 2);
  bf16_t* ipwb   = (bf16_t*)allocb((size_t)LL * 2 * DI * HH * 2);
  bf16_t* xpwb   = (bf16_t*)allocb((size_t)LL * 64 * DI * 2);
  bf16_t* dtwb   = (bf16_t*)allocb((size_t)LL * DI * RR * 2);
  bf16_t* owb    = (bf16_t*)allocb((size_t)LL * HH * DI * 2);
  bf16_t* outwb  = (bf16_t*)allocb((size_t)OUTF * HH * 2);

  auto cvt = [&](const float* src, bf16_t* dst, int n) {
    convert_bf16_kernel<<<(n + 255) / 256, 256, 0, stream>>>(src, dst, n);
  };
  cvt(in_w,    in_wb,  HH * INF_);
  cvt(in_proj, ipwb,   LL * 2 * DI * HH);
  cvt(x_proj,  xpwb,   LL * 64 * DI);
  cvt(dt_w,    dtwb,   LL * DI * RR);
  cvt(mb_out,  owb,    LL * HH * DI);
  cvt(out_w,   outwb,  OUTF * HH);

  // 1) NaN row mask + bf16 x
  rowmask_conv_kernel<<<NTOK / 8, 256, 0, stream>>>(x, mask, xb);

  // 2) input projection: h = (nan_to_num(x) @ in_w^T + in_b) * mask
  gemm_bf16_wmma<2, false><<<dim3(NTOK / 64, HH / 64), 256, 0, stream>>>(
      xb, INF_, in_wb, INF_, h, HH, nullptr, INF_, in_b, mask);

  for (int l = 0; l < LL; ++l) {
    const bf16_t* ipw = ipwb + (size_t)l * 2 * DI * HH;
    const float*  cw  = conv_w + (size_t)l * DI * KK;
    const float*  cb  = conv_b + (size_t)l * DI;
    const bf16_t* xpw = xpwb + (size_t)l * 64 * DI;
    const bf16_t* dtw = dtwb + (size_t)l * DI * RR;
    const float*  dtb = dt_b + (size_t)l * DI;
    const float*  Al  = A_log + (size_t)l * DI * SS;
    const float*  Dpl = D_ssm + (size_t)l * DI;
    const bf16_t* ow  = owb + (size_t)l * HH * DI;

    // RMSNorm -> bf16
    rmsnorm_kernel<<<NTOK / 8, 256, 0, stream>>>(h, rms_w + (size_t)l * HH, xnb);

    // xz = xn @ in_proj^T   (4096 x 2048, K=512)
    gemm_bf16_wmma<0, false><<<dim3(NTOK / 64, (2 * DI) / 64), 256, 0, stream>>>(
        xnb, HH, ipw, HH, xz, 2 * DI, nullptr, HH, nullptr, nullptr);

    // depthwise causal conv + SiLU -> u (f32) and ub (bf16)
    conv_silu_kernel<<<dim3(NTOK, DI / 256), 256, 0, stream>>>(xz, cw, cb, u, ub);

    // xd = u @ x_proj^T     (4096 x 64, K=1024), dual f32+bf16 output
    gemm_bf16_wmma<0, true><<<dim3(NTOK / 64, 64 / 64), 256, 0, stream>>>(
        ub, DI, xpw, DI, xd, 64, xdb, DI, nullptr, nullptr);

    // delta = softplus(dt @ dt_proj^T + dt_b)  (4096 x 1024, K=32; dt = xdb[:, :32])
    gemm_bf16_wmma<3, false><<<dim3(NTOK / 64, DI / 64), 256, 0, stream>>>(
        xdb, 64, dtw, RR, delta, DI, nullptr, RR, dtb, nullptr);

    // selective scan + D skip + SiLU(z) gating -> bf16 yg
    scan_kernel<<<BB * (DI / 256), 256, 0, stream>>>(delta, u, xd, xz, Al, Dpl, ygb);

    // residual: h += yg @ mb_out^T   (4096 x 512, K=1024)
    gemm_bf16_wmma<4, false><<<dim3(NTOK / 64, HH / 64), 256, 0, stream>>>(
        ygb, DI, ow, DI, h, HH, nullptr, DI, nullptr, nullptr);
  }

  // LayerNorm -> bf16
  layernorm_kernel<<<NTOK / 8, 256, 0, stream>>>(h, ln_g, ln_b, hnb);

  // out = hn @ out_w^T + out_b   (4096 x 256, K=512)
  gemm_bf16_wmma<1, false><<<dim3(NTOK / 64, OUTF / 64), 256, 0, stream>>>(
      hnb, HH, outwb, HH, outp, OUTF, nullptr, HH, out_b, nullptr);
}
